// HyperComplexAdapter_64149631533647
// MI455X (gfx1250) — compile-verified, matrix operand order checked by reading, not run on hardware
//
#include <hip/hip_runtime.h>
#include <hip/hip_bf16.h>

// ---------------------------------------------------------------------------
// PHM adapter for MI455X (gfx1250):
//   Wd = kron(phm, W_down)  [2048 x 512]   (precomputed transposed, bf16)
//   Wu = kron(phm, W_up)    [512  x 2048]  (precomputed transposed, bf16)
//   z   = relu(x @ Wd + b_down)   -> bf16 workspace
//   out = z @ Wu + b_up           -> f32 d_out
// GEMMs use v_wmma_f32_16x16x32_bf16 (wave32), double-buffered LDS A-tiles
// (one barrier per 32-deep K-step, global loads overlapped with WMMAs).
// ---------------------------------------------------------------------------

typedef __attribute__((ext_vector_type(16))) __bf16 v16bf;
typedef __attribute__((ext_vector_type(8)))  __bf16 v8bf;
typedef __attribute__((ext_vector_type(8)))  float  v8f;
typedef __attribute__((ext_vector_type(4)))  float  v4f;

#define N_PHM 4

// ---------------------------------------------------------------------------
// Kron-weight expansion into TRANSPOSED bf16 layout:
//   WT[o, k] = Wfull[k, o] = sum_i phm[i, p, r] * W[i, q, s]
// where k = p*qdim + q (input dim), o = r*sdim + s (output dim).
// ---------------------------------------------------------------------------
__global__ __launch_bounds__(256)
void phm_expand_kron(const float* __restrict__ phm, const float* __restrict__ W,
                     __bf16* __restrict__ WT, int in_f, int qdim, int sdim, int total)
{
    int idx = blockIdx.x * 256 + threadIdx.x;
    if (idx >= total) return;
    int k = idx % in_f;
    int o = idx / in_f;
    int p = k / qdim, q = k - p * qdim;
    int r = o / sdim, s = o - r * sdim;
    float acc = 0.0f;
#pragma unroll
    for (int i = 0; i < N_PHM; ++i) {
        acc += phm[i * 16 + p * 4 + r] * W[((size_t)i * qdim + q) * sdim + s];
    }
    WT[(size_t)o * in_f + k] = (__bf16)acc;
}

// ---------------------------------------------------------------------------
// Tiled GEMM:  Out[M,N] = act(A[M,K] @ WT[N,K]^T + bias[N])
//   A_F32:    A is f32 in global (converted to bf16 while staging to LDS),
//             else A is bf16 in global.
// Block: 256 thr = 8 waves; tile 128(M) x 128(N); wave = 64x32 (4x2 WMMA tiles).
// Double-buffered LDS A-tile; requires K % 64 == 0.
// ---------------------------------------------------------------------------
#define LDS_STRIDE 40   // 32 bf16 + pad; 80 B rows -> 16B aligned, bank-spread

template <bool A_F32, bool RELU, bool OUT_BF16>
__global__ __launch_bounds__(256)
void phm_gemm_bf16(const void* __restrict__ Av, const __bf16* __restrict__ WT,
                   const float* __restrict__ bias, void* __restrict__ Outv,
                   int M, int N, int K)
{
    __shared__ __bf16 smem0[128 * LDS_STRIDE];   // 10 KB A-tile buffer 0
    __shared__ __bf16 smem1[128 * LDS_STRIDE];   // 10 KB A-tile buffer 1

    const int m0 = blockIdx.x * 128;
    const int n0 = blockIdx.y * 128;

    const int tid  = threadIdx.x;
    const int lane = tid & 31;
    const int w    = tid >> 5;          // wave id 0..7
    const int wm   = w >> 2;            // 0..1  (M direction, 64 rows each)
    const int wn   = w & 3;             // 0..3  (N direction, 32 cols each)
    const int h    = (lane >> 4) & 1;   // lane half (K-split per ISA layout)
    const int l15  = lane & 15;

    // staging coordinates: 2 threads per row, 16 elements each
    const int sr = tid >> 1;
    const int sk = (tid & 1) * 16;

    // per-lane output columns + bias for the 2 N-tiles of this wave
    int   ncol[2];
    float bcol[2];
#pragma unroll
    for (int nt = 0; nt < 2; ++nt) {
        ncol[nt] = n0 + wn * 32 + nt * 16 + l15;
        bcol[nt] = bias[ncol[nt]];
    }

    v8f acc[4][2];
#pragma unroll
    for (int mt = 0; mt < 4; ++mt)
#pragma unroll
        for (int nt = 0; nt < 2; ++nt)
            acc[mt][nt] = (v8f){0.f, 0.f, 0.f, 0.f, 0.f, 0.f, 0.f, 0.f};

    // register-staged A tile (16 bf16 per thread)
    v8bf st0, st1;

    // ---- load one 128x32 A sub-tile (global -> regs, with f32->bf16 cvt) ----
    auto stage_load = [&](int kk) {
        if (A_F32) {
            const float* src = (const float*)Av + (size_t)(m0 + sr) * K + kk + sk;
            v4f f0 = *(const v4f*)(src + 0);
            v4f f1 = *(const v4f*)(src + 4);
            v4f f2 = *(const v4f*)(src + 8);
            v4f f3 = *(const v4f*)(src + 12);
#pragma unroll
            for (int j = 0; j < 4; ++j) { st0[j] = (__bf16)f0[j]; st0[j + 4] = (__bf16)f1[j]; }
#pragma unroll
            for (int j = 0; j < 4; ++j) { st1[j] = (__bf16)f2[j]; st1[j + 4] = (__bf16)f3[j]; }
        } else {
            const __bf16* src = (const __bf16*)Av + (size_t)(m0 + sr) * K + kk + sk;
            st0 = *(const v8bf*)(src + 0);
            st1 = *(const v8bf*)(src + 8);
        }
    };

    // ---- regs -> LDS buffer ----
    auto stage_store = [&](__bf16* buf) {
        *(v8bf*)&buf[sr * LDS_STRIDE + sk + 0] = st0;
        *(v8bf*)&buf[sr * LDS_STRIDE + sk + 8] = st1;
    };

    // ---- one 32-deep K-step: fragments from LDS/global, 8 WMMAs ----
    auto compute = [&](const __bf16* buf, int kk) {
        v16bf afrag[4];
#pragma unroll
        for (int mt = 0; mt < 4; ++mt) {
            const __bf16* p = &buf[(wm * 64 + mt * 16 + l15) * LDS_STRIDE + h * 8];
            v8bf lo = *(const v8bf*)(p + 0);     // K = h*8 .. h*8+7
            v8bf hi = *(const v8bf*)(p + 16);    // K = 16 + h*8 ..
            afrag[mt] = __builtin_shufflevector(lo, hi, 0, 1, 2, 3, 4, 5, 6, 7,
                                                        8, 9, 10, 11, 12, 13, 14, 15);
        }
        v16bf bfrag[2];
#pragma unroll
        for (int nt = 0; nt < 2; ++nt) {
            const __bf16* p = WT + (size_t)ncol[nt] * K + kk + h * 8;
            v8bf lo = *(const v8bf*)(p + 0);
            v8bf hi = *(const v8bf*)(p + 16);
            bfrag[nt] = __builtin_shufflevector(lo, hi, 0, 1, 2, 3, 4, 5, 6, 7,
                                                        8, 9, 10, 11, 12, 13, 14, 15);
        }
#pragma unroll
        for (int mt = 0; mt < 4; ++mt)
#pragma unroll
            for (int nt = 0; nt < 2; ++nt)
                acc[mt][nt] = __builtin_amdgcn_wmma_f32_16x16x32_bf16(
                    false, afrag[mt], false, bfrag[nt],
                    (short)0, acc[mt][nt], false, false);
    };

    // ---- software pipeline: double-buffered, one barrier per K-step ----
    stage_load(0);
    stage_store(smem0);
    __syncthreads();

    for (int k0 = 0; k0 < K; k0 += 64) {
        // step A: compute on smem0, prefetch k0+32 into regs, fill smem1
        stage_load(k0 + 32);                 // global loads overlap WMMAs below
        compute(smem0, k0);
        stage_store(smem1);
        __syncthreads();

        // step B: compute on smem1, prefetch k0+64 into regs, fill smem0
        const bool more = (k0 + 64) < K;
        if (more) stage_load(k0 + 64);
        compute(smem1, k0 + 32);
        if (more) stage_store(smem0);
        __syncthreads();
    }

    // ---- epilogue: bias (+relu), store per C/D layout (M = i + 8h, N = l15) ----
#pragma unroll
    for (int mt = 0; mt < 4; ++mt) {
#pragma unroll
        for (int nt = 0; nt < 2; ++nt) {
#pragma unroll
            for (int i = 0; i < 8; ++i) {
                int row = m0 + wm * 64 + mt * 16 + h * 8 + i;
                float v = acc[mt][nt][i] + bcol[nt];
                if (RELU) v = fmaxf(v, 0.0f);
                if (OUT_BF16)
                    ((__bf16*)Outv)[(size_t)row * N + ncol[nt]] = (__bf16)v;
                else
                    ((float*)Outv)[(size_t)row * N + ncol[nt]] = v;
            }
        }
    }
}

// ---------------------------------------------------------------------------
extern "C" void kernel_launch(void* const* d_in, const int* in_sizes, int n_in,
                              void* d_out, int out_size, void* d_ws, size_t ws_size,
                              hipStream_t stream)
{
    const float* x    = (const float*)d_in[0];   // [8192, 2048]
    const float* phm  = (const float*)d_in[1];   // [4, 4, 4]
    const float* W_dn = (const float*)d_in[2];   // [4, 512, 128]
    const float* b_dn = (const float*)d_in[3];   // [512]
    const float* W_up = (const float*)d_in[4];   // [4, 128, 512]
    const float* b_up = (const float*)d_in[5];   // [2048]

    const int T = 8192, D = 2048, B = 512;

    // workspace layout (bf16): WdT [B x D], WuT [D x B], z [T x B]  = 12 MB
    __bf16* wdT = (__bf16*)d_ws;
    __bf16* wuT = wdT + (size_t)B * D;
    __bf16* z   = wuT + (size_t)D * B;

    // 1) expand kron weights (transposed, bf16)
    {
        int total = B * D;                    // WdT: out_f=B, in_f=D
        phm_expand_kron<<<(total + 255) / 256, 256, 0, stream>>>(
            phm, W_dn, wdT, /*in_f=*/D, /*qdim=*/D / 4, /*sdim=*/B / 4, total);
    }
    {
        int total = D * B;                    // WuT: out_f=D, in_f=B
        phm_expand_kron<<<(total + 255) / 256, 256, 0, stream>>>(
            phm, W_up, wuT, /*in_f=*/B, /*qdim=*/B / 4, /*sdim=*/D / 4, total);
    }

    // 2) z = relu(x @ Wd + b_down)   [f32 in, bf16 out]
    {
        dim3 grid(T / 128, B / 128);          // 64 x 4
        phm_gemm_bf16<true, true, true><<<grid, 256, 0, stream>>>(
            (const void*)x, wdT, b_dn, (void*)z, T, B, D);
    }

    // 3) out = z @ Wu + b_up         [bf16 in, f32 out]
    {
        dim3 grid(T / 128, D / 128);          // 64 x 16
        phm_gemm_bf16<false, false, false><<<grid, 256, 0, stream>>>(
            (const void*)z, wuT, b_up, d_out, T, D, B);
    }
}
